// Attention_1992864825947
// MI455X (gfx1250) — matrix operand order repaired and checked by reading, not compiled, not run on hardware
//
#include <hip/hip_runtime.h>
#include <hip/hip_bf16.h>

// ---------------- problem constants ----------------
#define NUM_HIDDEN 768
#define NUM_HEADS  4
#define D_HEAD     192          // 768/4
#define BATCH      8
#define SEQ        1024
#define HB         (NUM_HEADS * BATCH)   // 32 head-batches
#define MROWS      (BATCH * SEQ)         // 8192
#define MASK_FILL  (-4294967295.0f)      // -(2^32-1)
#define LN_EPS     1e-5f

// ---------------- CDNA5 WMMA types ----------------
typedef __bf16 bf16;
typedef bf16  v16bf __attribute__((ext_vector_type(16)));
typedef float v8f   __attribute__((ext_vector_type(8)));

union FragU { v16bf v; uint4 q[2]; };

__device__ __forceinline__ v8f vzero8() {
  v8f r;
#pragma unroll
  for (int i = 0; i < 8; ++i) r[i] = 0.0f;
  return r;
}

__device__ __forceinline__ unsigned short f32_to_bf16(float f) {
  unsigned int u = __float_as_uint(f);
  u = u + 0x7FFFu + ((u >> 16) & 1u);   // round-to-nearest-even
  return (unsigned short)(u >> 16);
}

// A-matrix fragment (16x32 bf16).  ISA 7.12.2: lane m = L%16; g = L/16;
// elements e<8 -> k = 8g+e ; e>=8 -> k = 16+8g+(e-8).
__device__ __forceinline__ v16bf load_frag_a(const unsigned short* p, int g) {
  FragU f;
  f.q[0] = *(const uint4*)(p + 8 * g);
  f.q[1] = *(const uint4*)(p + 16 + 8 * g);
  return f.v;
}

// B-matrix fragment (32x16 bf16).  lane n = L%16; g = L/16;
// elements e<8 -> k = 8g+e ; e>=8 -> k = 16+8g+(e-8).
__device__ __forceinline__ v16bf load_frag_b(const unsigned short* p, int g) {
  FragU f;
  f.q[0] = *(const uint4*)(p + 8 * g);
  f.q[1] = *(const uint4*)(p + 16 + 8 * g);
  return f.v;
}

__device__ __forceinline__ v8f wmma_bf16(v16bf a, v16bf b, v8f c) {
  return __builtin_amdgcn_wmma_f32_16x16x32_bf16(false, a, false, b,
                                                 (short)0, c, false, false);
}

// ---------------- CDNA5 async global->LDS copy (ASYNCcnt) ----------------
// 16-byte per-lane async copy; the builtin takes pointers to 16B int
// vectors in AS1 (global) / AS3 (LDS).  Falls back to a synchronous copy
// if the builtin is not available on this toolchain.
__device__ __forceinline__ void async_copy16(const void* g, void* l) {
#if defined(__AMDGCN__) && __has_builtin(__builtin_amdgcn_global_load_async_to_lds_b128)
  typedef int v4i_ __attribute__((vector_size(16)));
  typedef __attribute__((address_space(1))) v4i_* gv4p;
  typedef __attribute__((address_space(3))) v4i_* lv4p;
  __builtin_amdgcn_global_load_async_to_lds_b128((gv4p)(void*)g, (lv4p)l, 0, 0);
#else
  *(uint4*)l = *(const uint4*)g;
#endif
}

__device__ __forceinline__ void async_wait_all() {
#if defined(__AMDGCN__) && __has_builtin(__builtin_amdgcn_global_load_async_to_lds_b128)
  asm volatile("s_wait_asynccnt 0x0" ::: "memory");
#endif
}

// =====================================================================
// Kernel E: f32 -> bf16 pre-conversion (bandwidth-bound, enables pure
// async-LDS staging in the FF GEMM).  4 elems/thread, vectorized.
// =====================================================================
__global__ __launch_bounds__(256)
void bf16_convert_kernel(const float* __restrict__ src,
                         unsigned short* __restrict__ dst, int n) {
  int i = blockIdx.x * 1024 + threadIdx.x * 4;
  if (i + 3 < n) {
    float4 v = *(const float4*)(src + i);
    union { unsigned short u[4]; uint2 q; } pk;
    pk.u[0] = f32_to_bf16(v.x);
    pk.u[1] = f32_to_bf16(v.y);
    pk.u[2] = f32_to_bf16(v.z);
    pk.u[3] = f32_to_bf16(v.w);
    *(uint2*)(dst + i) = pk.q;
  }
}

// =====================================================================
// Kernel A: QKV projections.  C = X (8192x768) * W^T (768x768)
// z=0: K (bf16 [hb][s][d]) ; z=1: V TRANSPOSED (bf16 [hb][d][s]) ;
// z=2: Q (bf16 [hb][s][d]).  Tile 64x128, BK=32, 8 waves * 32x32.
// =====================================================================
__global__ __launch_bounds__(256)
void qkv_proj_kernel(const float* __restrict__ memory,
                     const float* __restrict__ decoder,
                     const float* __restrict__ WK,
                     const float* __restrict__ WV,
                     const float* __restrict__ WQ,
                     unsigned short* __restrict__ Kbf,
                     unsigned short* __restrict__ Vt,
                     unsigned short* __restrict__ Qbf) {
  __shared__ unsigned short As[64 * 32];    // [m][k]
  __shared__ unsigned short Bs[128 * 32];   // [n][k]

  const int z = blockIdx.z;
  const float* X = (z == 2) ? decoder : memory;
  const float* W = (z == 0) ? WK : (z == 1) ? WV : WQ;

  const int m0 = blockIdx.y * 64;
  const int n0 = blockIdx.x * 128;
  const int tid = threadIdx.x;
  const int lane = tid & 31, wave = tid >> 5;
  const int wm = wave >> 2, wn = wave & 3;      // 2x4 wave grid
  const int g = lane >> 4, ln = lane & 15;

  const int arow = tid >> 2, akseg = (tid & 3) * 8;
  const int brow = tid >> 1, bkseg = (tid & 1) * 16;

  v8f acc[2][2];
#pragma unroll
  for (int i = 0; i < 2; ++i)
#pragma unroll
    for (int j = 0; j < 2; ++j) acc[i][j] = vzero8();

  for (int k0 = 0; k0 < NUM_HIDDEN; k0 += 32) {
#pragma unroll
    for (int i = 0; i < 8; ++i) {
      float v = X[(size_t)(m0 + arow) * NUM_HIDDEN + k0 + akseg + i];
      As[arow * 32 + akseg + i] = f32_to_bf16(v);
    }
#pragma unroll
    for (int i = 0; i < 16; ++i) {
      float v = W[(size_t)(n0 + brow) * NUM_HIDDEN + k0 + bkseg + i];
      Bs[brow * 32 + bkseg + i] = f32_to_bf16(v);
    }
    __syncthreads();

    v16bf a0 = load_frag_a(&As[(wm * 32 + ln) * 32], g);
    v16bf a1 = load_frag_a(&As[(wm * 32 + 16 + ln) * 32], g);
    v16bf b0 = load_frag_b(&Bs[(wn * 32 + ln) * 32], g);
    v16bf b1 = load_frag_b(&Bs[(wn * 32 + 16 + ln) * 32], g);
    acc[0][0] = wmma_bf16(a0, b0, acc[0][0]);
    acc[0][1] = wmma_bf16(a0, b1, acc[0][1]);
    acc[1][0] = wmma_bf16(a1, b0, acc[1][0]);
    acc[1][1] = wmma_bf16(a1, b1, acc[1][1]);
    __syncthreads();
  }

  // epilogue: one top-level branch on z; h/d hoisted per column tile.
  const int b = m0 >> 10;   // 64-row tile lies within one batch
  if (z == 1) {
    // Vt[hb][d][s]: the 8 accumulator rows are contiguous s -> 16B store
#pragma unroll
    for (int mi = 0; mi < 2; ++mi) {
      const int s_base = (m0 + wm * 32 + mi * 16 + 8 * g) & 1023;
#pragma unroll
      for (int ni = 0; ni < 2; ++ni) {
        const int n = n0 + wn * 32 + ni * 16 + ln;
        const int h = n / D_HEAD, d = n % D_HEAD;
        const int hb = h * BATCH + b;
        union { unsigned short u[8]; uint4 q; } pk;
#pragma unroll
        for (int j = 0; j < 8; ++j) pk.u[j] = f32_to_bf16(acc[mi][ni][j]);
        *(uint4*)&Vt[((size_t)hb * D_HEAD + d) * SEQ + s_base] = pk.q;
      }
    }
  } else {
    unsigned short* dst = (z == 0) ? Kbf : Qbf;
#pragma unroll
    for (int mi = 0; mi < 2; ++mi) {
      const int s_base = (m0 + wm * 32 + mi * 16 + 8 * g) & 1023;
#pragma unroll
      for (int ni = 0; ni < 2; ++ni) {
        const int n = n0 + wn * 32 + ni * 16 + ln;
        const int h = n / D_HEAD, d = n % D_HEAD;
        const int hb = h * BATCH + b;
        unsigned short* p = dst + ((size_t)hb * SEQ + s_base) * D_HEAD + d;
#pragma unroll
        for (int j = 0; j < 8; ++j)
          p[(size_t)j * D_HEAD] = f32_to_bf16(acc[mi][ni][j]);
      }
    }
  }
}

// =====================================================================
// Kernel B: attention for one (hb, 32 q-rows) block.
//   stage Q tile (async -> LDS), phase 1: S = Q*K^T*scale -> LDS f32
//   mask pass (coalesced uchar4), phase 2: softmax + query-mask
//     -> d_out f32 probs + LDS bf16 probs, phase 3: O = P*V -> attnO.
// 256 threads = 8 waves; dynamic LDS = 204KB (CDNA5: 320KB/WGP).
// =====================================================================
__global__ __launch_bounds__(256)
void attention_kernel(const unsigned short* __restrict__ Qbf,
                      const unsigned short* __restrict__ Kbf,
                      const unsigned short* __restrict__ Vt,
                      const unsigned char* __restrict__ mask,
                      const float* __restrict__ qmask,
                      float* __restrict__ attnOut,          // [32][1024][1024]
                      unsigned short* __restrict__ attnO) { // [8192][768] bf16
  extern __shared__ unsigned char smem[];
  float* sS = (float*)smem;                                     // 32x1024 f32
  unsigned short* sP = (unsigned short*)(smem + 32 * 1024 * 4); // 32x1024 bf16
  unsigned short* sQ =
      (unsigned short*)(smem + 32 * 1024 * 4 + 32 * 1024 * 2);  // 32x192 bf16

  const int q0 = blockIdx.x * 32;
  const int hb = blockIdx.y;
  const int h = hb >> 3, b = hb & 7;
  const int tid = threadIdx.x, lane = tid & 31, wave = tid >> 5;
  const int g = lane >> 4, ln = lane & 15;
  const float scale = 0.07216878364870322f;   // 1/sqrt(192)

  const unsigned short* Qh = Qbf + (size_t)hb * SEQ * D_HEAD;
  const unsigned short* Kh = Kbf + (size_t)hb * SEQ * D_HEAD;
  const unsigned short* Vh = Vt + (size_t)hb * D_HEAD * SEQ;

  // ---- stage Q tile: 32x192 bf16 = 768 x 16B async transfers ----
#pragma unroll
  for (int i = 0; i < 3; ++i) {
    int t = tid + 256 * i;
    int row = t / 24, seg = (t % 24) * 8;
    async_copy16(Qh + (size_t)(q0 + row) * D_HEAD + seg, &sQ[row * 192 + seg]);
  }
  async_wait_all();
  __syncthreads();

  // ---- phase 1: each wave does kv block [wave*128, wave*128+128) ----
  const int kvb = wave * 128;
  v8f acc[2][8];
#pragma unroll
  for (int i = 0; i < 2; ++i)
#pragma unroll
    for (int j = 0; j < 8; ++j) acc[i][j] = vzero8();

  for (int d0 = 0; d0 < D_HEAD; d0 += 32) {
    v16bf a0 = load_frag_a(&sQ[ln * 192 + d0], g);
    v16bf a1 = load_frag_a(&sQ[(16 + ln) * 192 + d0], g);
#pragma unroll
    for (int ni = 0; ni < 8; ++ni) {
      v16bf bk = load_frag_b(Kh + (size_t)(kvb + ni * 16 + ln) * D_HEAD + d0, g);
      acc[0][ni] = wmma_bf16(a0, bk, acc[0][ni]);
      acc[1][ni] = wmma_bf16(a1, bk, acc[1][ni]);
    }
  }

#pragma unroll
  for (int mi = 0; mi < 2; ++mi)
#pragma unroll
    for (int ni = 0; ni < 8; ++ni)
#pragma unroll
      for (int j = 0; j < 8; ++j) {
        int qloc = mi * 16 + j + 8 * g;
        int kv = kvb + ni * 16 + ln;
        sS[qloc * 1024 + kv] = acc[mi][ni][j] * scale;
      }
  __syncthreads();

  // ---- mask pass: coalesced uchar4 reads, 4 elems/thread/iter ----
  {
    const unsigned char* mrow = mask + ((size_t)b * SEQ + q0) * SEQ;
#pragma unroll
    for (int t = 0; t < 32; ++t) {
      int idx = (tid + 256 * t) * 4;             // 0..32764, row-major 32x1024
      uchar4 mk = *(const uchar4*)(mrow + idx);  // mask[b][q0+row][col..col+3]
      if (mk.x) sS[idx + 0] = MASK_FILL;
      if (mk.y) sS[idx + 1] = MASK_FILL;
      if (mk.z) sS[idx + 2] = MASK_FILL;
      if (mk.w) sS[idx + 3] = MASK_FILL;
    }
  }
  __syncthreads();

  // ---- phase 2: softmax; each wave handles 4 rows ----
  for (int i = 0; i < 4; ++i) {
    int r = wave * 4 + i;
    float mx = -3.0e38f;
#pragma unroll
    for (int t = 0; t < 32; ++t) mx = fmaxf(mx, sS[r * 1024 + lane + 32 * t]);
#pragma unroll
    for (int off = 16; off > 0; off >>= 1) mx = fmaxf(mx, __shfl_xor(mx, off, 32));
    float sum = 0.0f;
#pragma unroll
    for (int t = 0; t < 32; ++t) sum += __expf(sS[r * 1024 + lane + 32 * t] - mx);
#pragma unroll
    for (int off = 16; off > 0; off >>= 1) sum += __shfl_xor(sum, off, 32);
    float inv = qmask[b * SEQ + q0 + r] / sum;   // post-softmax query-mask
#pragma unroll
    for (int t = 0; t < 32; ++t) {
      int c = lane + 32 * t;
      float p = __expf(sS[r * 1024 + c] - mx) * inv;
      attnOut[((size_t)hb * SEQ + q0 + r) * SEQ + c] = p;
      sP[r * 1024 + c] = f32_to_bf16(p);
    }
  }
  __syncthreads();

  // ---- phase 3: O(32x192) = P(32x1024) * V(1024x192) ----
  const int nni = (wave < 4) ? 2 : 1;    // 12 n-tiles over 8 waves
  v8f accO[2][2];
#pragma unroll
  for (int i = 0; i < 2; ++i)
#pragma unroll
    for (int j = 0; j < 2; ++j) accO[i][j] = vzero8();

  for (int kv0 = 0; kv0 < SEQ; kv0 += 32) {
    v16bf a0 = load_frag_a(&sP[ln * 1024 + kv0], g);
    v16bf a1 = load_frag_a(&sP[(16 + ln) * 1024 + kv0], g);
#pragma unroll
    for (int u = 0; u < 2; ++u) {
      if (u < nni) {
        int ni = wave + u * 8;
        v16bf bv = load_frag_b(Vh + (size_t)(ni * 16 + ln) * SEQ + kv0, g);
        accO[0][u] = wmma_bf16(a0, bv, accO[0][u]);
        accO[1][u] = wmma_bf16(a1, bv, accO[1][u]);
      }
    }
  }

  for (int u = 0; u < nni; ++u) {
    int ni = wave + u * 8;
    const int d = ni * 16 + ln;
#pragma unroll
    for (int mi = 0; mi < 2; ++mi) {
      const int qb = q0 + mi * 16 + 8 * g;
      unsigned short* p =
          attnO + ((size_t)b * SEQ + qb) * NUM_HIDDEN + h * D_HEAD + d;
#pragma unroll
      for (int j = 0; j < 8; ++j)
        p[(size_t)j * NUM_HIDDEN] = f32_to_bf16(accO[mi][u][j]);
    }
  }
}

// =====================================================================
// Kernel C: FF GEMM.  C[8192x768] = concat(dec, attn)[8192x1536]*Wf^T + bf
// Both operands pre-converted to bf16 -> pure async global->LDS staging.
// =====================================================================
__global__ __launch_bounds__(256)
void ff_gemm_kernel(const unsigned short* __restrict__ decbf,
                    const unsigned short* __restrict__ attnO,
                    const unsigned short* __restrict__ Wfbf,
                    const float* __restrict__ bfv,
                    float* __restrict__ ffout) {
  __shared__ unsigned short As[64 * 32];
  __shared__ unsigned short Bs[128 * 32];

  const int m0 = blockIdx.y * 64;
  const int n0 = blockIdx.x * 128;
  const int tid = threadIdx.x;
  const int lane = tid & 31, wave = tid >> 5;
  const int wm = wave >> 2, wn = wave & 3;
  const int g = lane >> 4, ln = lane & 15;

  const int arow = tid >> 2, aseg = (tid & 3) * 8;   // A: 1 x 16B / thread

  v8f acc[2][2];
#pragma unroll
  for (int i = 0; i < 2; ++i)
#pragma unroll
    for (int j = 0; j < 2; ++j) acc[i][j] = vzero8();

  for (int k0 = 0; k0 < 2 * NUM_HIDDEN; k0 += 32) {
    // A tile: first half from decbf, second half from attnO (both bf16)
    const unsigned short* Asrc = (k0 < NUM_HIDDEN) ? decbf : attnO;
    const int gk0 = (k0 < NUM_HIDDEN) ? k0 : k0 - NUM_HIDDEN;
    async_copy16(Asrc + (size_t)(m0 + arow) * NUM_HIDDEN + gk0 + aseg,
                 &As[arow * 32 + aseg]);
    // B tile: 128 rows x 32 k from Wfbf  (2 x 16B / thread)
#pragma unroll
    for (int i = 0; i < 2; ++i) {
      int t = tid + 256 * i;
      int row = t >> 2, seg = (t & 3) * 8;
      async_copy16(Wfbf + (size_t)(n0 + row) * (2 * NUM_HIDDEN) + k0 + seg,
                   &Bs[row * 32 + seg]);
    }
    async_wait_all();
    __syncthreads();

    v16bf a0 = load_frag_a(&As[(wm * 32 + ln) * 32], g);
    v16bf a1 = load_frag_a(&As[(wm * 32 + 16 + ln) * 32], g);
    v16bf b0 = load_frag_b(&Bs[(wn * 32 + ln) * 32], g);
    v16bf b1 = load_frag_b(&Bs[(wn * 32 + 16 + ln) * 32], g);
    acc[0][0] = wmma_bf16(a0, b0, acc[0][0]);
    acc[0][1] = wmma_bf16(a0, b1, acc[0][1]);
    acc[1][0] = wmma_bf16(a1, b0, acc[1][0]);
    acc[1][1] = wmma_bf16(a1, b1, acc[1][1]);
    __syncthreads();
  }

#pragma unroll
  for (int mi = 0; mi < 2; ++mi)
#pragma unroll
    for (int ni = 0; ni < 2; ++ni) {
      const int n = n0 + wn * 32 + ni * 16 + ln;
      const float bias = bfv[n];
      const int mb = m0 + wm * 32 + mi * 16 + 8 * g;
#pragma unroll
      for (int j = 0; j < 8; ++j)
        ffout[(size_t)(mb + j) * NUM_HIDDEN + n] = acc[mi][ni][j] + bias;
    }
}

// =====================================================================
// Kernel D: residual + LayerNorm.  One wave per 768-elem row.
// =====================================================================
__global__ __launch_bounds__(256)
void layernorm_kernel(const float* __restrict__ ffout,
                      const float* __restrict__ decoder,
                      const float* __restrict__ gamma,
                      const float* __restrict__ beta,
                      float* __restrict__ out) {
  const int wave = threadIdx.x >> 5, lane = threadIdx.x & 31;
  const int row = blockIdx.x * 8 + wave;
  const float* fr = ffout + (size_t)row * NUM_HIDDEN;
  const float* dr = decoder + (size_t)row * NUM_HIDDEN;

  float x[24];
  float s = 0.0f, s2 = 0.0f;
#pragma unroll
  for (int t = 0; t < 24; ++t) {
    x[t] = fr[lane + 32 * t] + dr[lane + 32 * t];   // residual
    s += x[t];
    s2 += x[t] * x[t];
  }
#pragma unroll
  for (int off = 16; off > 0; off >>= 1) {
    s += __shfl_xor(s, off, 32);
    s2 += __shfl_xor(s2, off, 32);
  }
  const float mu = s * (1.0f / NUM_HIDDEN);
  const float var = s2 * (1.0f / NUM_HIDDEN) - mu * mu;
  const float inv = rsqrtf(var + LN_EPS);
#pragma unroll
  for (int t = 0; t < 24; ++t) {
    int c = lane + 32 * t;
    out[(size_t)row * NUM_HIDDEN + c] = (x[t] - mu) * inv * gamma[c] + beta[c];
  }
}

// =====================================================================
// Launcher.  d_in: memory, decoder_input, mask, query_mask,
//                  WK, WV, WQ, Wf, bf, gamma, beta
// d_out: result [8,1024,768] f32 then attention [8,4,1024,1024] f32
// =====================================================================
extern "C" void kernel_launch(void* const* d_in, const int* in_sizes, int n_in,
                              void* d_out, int out_size, void* d_ws,
                              size_t ws_size, hipStream_t stream) {
  const float* memory = (const float*)d_in[0];
  const float* decoder = (const float*)d_in[1];
  const unsigned char* mask = (const unsigned char*)d_in[2];  // jax bool = 1B
  const float* qmask = (const float*)d_in[3];
  const float* WK = (const float*)d_in[4];
  const float* WV = (const float*)d_in[5];
  const float* WQ = (const float*)d_in[6];
  const float* Wf = (const float*)d_in[7];
  const float* bfv = (const float*)d_in[8];
  const float* gamma = (const float*)d_in[9];
  const float* beta = (const float*)d_in[10];

  float* out_result = (float*)d_out;
  float* out_attn = (float*)d_out + (size_t)MROWS * NUM_HIDDEN;

  unsigned char* ws = (unsigned char*)d_ws;
  const size_t szHeadBF = (size_t)HB * SEQ * D_HEAD * 2;   // 12.58 MB
  const size_t szRowBF = (size_t)MROWS * NUM_HIDDEN * 2;   // 12.58 MB
  unsigned short* Kbf = (unsigned short*)(ws);
  unsigned short* Vt = (unsigned short*)(ws + szHeadBF);
  unsigned short* Qbf = (unsigned short*)(ws + 2 * szHeadBF);
  unsigned short* attnO = (unsigned short*)(ws + 3 * szHeadBF);
  unsigned short* decbf = (unsigned short*)(ws + 3 * szHeadBF + szRowBF);
  unsigned short* Wfbf = (unsigned short*)(ws + 3 * szHeadBF + 2 * szRowBF);
  float* ffout = (float*)(ws + 3 * szHeadBF + 2 * szRowBF +
                          (size_t)NUM_HIDDEN * 2 * NUM_HIDDEN * 2);

  bf16_convert_kernel<<<dim3(6144), 256, 0, stream>>>(
      decoder, decbf, MROWS * NUM_HIDDEN);
  bf16_convert_kernel<<<dim3(1152), 256, 0, stream>>>(
      Wf, Wfbf, NUM_HIDDEN * 2 * NUM_HIDDEN);

  qkv_proj_kernel<<<dim3(6, 128, 3), 256, 0, stream>>>(
      memory, decoder, WK, WV, WQ, Kbf, Vt, Qbf);

  attention_kernel<<<dim3(32, 32), 256, 208896, stream>>>(
      Qbf, Kbf, Vt, mask, qmask, out_attn, attnO);

  ff_gemm_kernel<<<dim3(6, 128), 256, 0, stream>>>(
      decbf, attnO, Wfbf, bfv, ffout);

  layernorm_kernel<<<dim3(1024), 256, 0, stream>>>(
      ffout, decoder, gamma, beta, out_result);
}
